// EDACrossIMUEncoder_1906965479428
// MI455X (gfx1250) — matrix operand order repaired and checked by reading, not compiled, not run on hardware
//
#include <hip/hip_runtime.h>

// Problem constants (from reference)
#define B_TOT 65536
#define D_IN  512
#define M_DIM 128
#define L_NUM 2

typedef _Float16 half_t;
typedef _Float16 h8  __attribute__((ext_vector_type(8)));
typedef _Float16 h16 __attribute__((ext_vector_type(16)));
typedef float    f32x8 __attribute__((ext_vector_type(8)));

// Per-wave LDS layout (bytes): eda16[16x128] imu16[16x128] h16[16x256] t16[16x128] stat[32f]
#define OFF_EDA  0
#define OFF_IMU  4096
#define OFF_H    8192
#define OFF_T    16384
#define OFF_STAT 20480
#define WAVE_BYTES 20736
#define WAVES_PER_BLK 8
#define SMEM_BYTES (WAVE_BYTES * WAVES_PER_BLK)

// ---------------------------------------------------------------------------
// WMMA tile loaders (wave32, V_WMMA_F32_16X16X32_F16 layouts, ISA 7.12.2)
// A (16x32 f16): lane -> row = L&15, hi = L>>4; halves 0..7 = K[k0+hi*8 ..],
//                halves 8..15 = K[k0+16+hi*8 ..]   (two 16B LDS b128 loads)
// B (32x16 f16): pre-swizzled in d_ws: tile = 32 lanes x 16 halves contiguous,
//                tiles ordered k-major within an n-tile row (one 32B load/lane)
// C/D (16x16 f32): N = L&15, M = v + 8*(L>>4)
// ---------------------------------------------------------------------------
__device__ __forceinline__ h16 load_a16(const half_t* A, int lda, int k0, int lane) {
    const half_t* p = A + (lane & 15) * lda + k0 + (lane >> 4) * 8;
    h8 lo = *(const h8*)p;
    h8 hi = *(const h8*)(p + 16);
    return __builtin_shufflevector(lo, hi, 0,1,2,3,4,5,6,7,8,9,10,11,12,13,14,15);
}

__device__ __forceinline__ h16 load_b16(const half_t* Wt, int tiles_k, int n0, int k0, int lane) {
    const h16* p = (const h16*)Wt;
    return p[(size_t)((n0 >> 4) * tiles_k + (k0 >> 5)) * 32 + lane];
}

// Out[16xN](f16 LDS) = A[16xK](f16 LDS) @ Wt^T + bias (opt relu)
template <int N, int K, bool RELU>
__device__ __forceinline__ void gemm_l(const half_t* A, int lda, const half_t* Wt,
                                       const float* bias, half_t* Out, int ldo, int lane) {
    constexpr int TK = K / 32;
    for (int n0 = 0; n0 < N; n0 += 16) {
        f32x8 c = {};
#pragma unroll 4
        for (int kt = 0; kt < TK; ++kt) {
            h16 a = load_a16(A, lda, kt * 32, lane);
            h16 b = load_b16(Wt, TK, n0, kt * 32, lane);
            c = __builtin_amdgcn_wmma_f32_16x16x32_f16(false, a, false, b,
                                                       (short)0, c, false, false);
        }
        int n = n0 + (lane & 15);
        float bv = bias[n];
#pragma unroll
        for (int v = 0; v < 8; ++v) {
            float val = c[v] + bv;
            if (RELU) val = fmaxf(val, 0.0f);
            Out[(v + 8 * (lane >> 4)) * ldo + n] = (half_t)val;
        }
    }
}

// Projection: A is fp32 tokens in HBM (read once), two bias vectors (bias + cls)
template <int K>
__device__ __forceinline__ void gemm_proj(const float* A, int lda, const half_t* Wt,
                                          const float* bias, const float* bias2,
                                          half_t* Out, int lane) {
    constexpr int TK = K / 32;
    for (int n0 = 0; n0 < 128; n0 += 16) {
        f32x8 c = {};
#pragma unroll 4
        for (int kt = 0; kt < TK; ++kt) {
            int k0 = kt * 32;
            const float* p = A + (size_t)(lane & 15) * lda + k0 + (lane >> 4) * 8;
            h16 a;
#pragma unroll
            for (int i = 0; i < 8; ++i) {
                a[i]     = (half_t)p[i];
                a[i + 8] = (half_t)p[i + 16];
            }
            h16 b = load_b16(Wt, TK, n0, k0, lane);
            c = __builtin_amdgcn_wmma_f32_16x16x32_f16(false, a, false, b,
                                                       (short)0, c, false, false);
        }
        int n = n0 + (lane & 15);
        float bv = bias[n] + bias2[n];
#pragma unroll
        for (int v = 0; v < 8; ++v)
            Out[(v + 8 * (lane >> 4)) * 128 + n] = (half_t)(c[v] + bv);
    }
}

// Out[16x128] = [A1 | A2](16x256) @ Wt(128x256)^T + bias   (fused concat head)
__device__ __forceinline__ void gemm_cat16(const half_t* A1, const half_t* A2,
                                           const half_t* Wt, const float* bias,
                                           half_t* Out, int lane) {
    for (int n0 = 0; n0 < 128; n0 += 16) {
        f32x8 c = {};
#pragma unroll 4
        for (int kt = 0; kt < 4; ++kt) {
            h16 a = load_a16(A1, 128, kt * 32, lane);
            h16 b = load_b16(Wt, 8, n0, kt * 32, lane);
            c = __builtin_amdgcn_wmma_f32_16x16x32_f16(false, a, false, b,
                                                       (short)0, c, false, false);
        }
#pragma unroll 4
        for (int kt = 0; kt < 4; ++kt) {
            h16 a = load_a16(A2, 128, kt * 32, lane);
            h16 b = load_b16(Wt, 8, n0, 128 + kt * 32, lane);
            c = __builtin_amdgcn_wmma_f32_16x16x32_f16(false, a, false, b,
                                                       (short)0, c, false, false);
        }
        int n = n0 + (lane & 15);
        float bv = bias[n];
#pragma unroll
        for (int v = 0; v < 8; ++v)
            Out[(v + 8 * (lane >> 4)) * 128 + n] = (half_t)(c[v] + bv);
    }
}

// X = LN(X + Y) * g + b over rows of 128 (f16 storage, f32 math)
__device__ __forceinline__ void residual_ln16(half_t* X, const half_t* Y,
                                              const float* g, const float* b,
                                              float* stat, int lane) {
    if (lane < 16) {
        float s = 0.0f, s2 = 0.0f;
#pragma unroll 8
        for (int j = 0; j < 128; ++j) {
            float v = (float)X[lane * 128 + j] + (float)Y[lane * 128 + j];
            s += v; s2 += v * v;
        }
        float m   = s * (1.0f / 128.0f);
        float var = s2 * (1.0f / 128.0f) - m * m;
        stat[lane]      = m;
        stat[16 + lane] = __frsqrt_rn(var + 1e-5f);
    }
    __syncthreads();
    for (int idx = lane; idx < 16 * 128; idx += 32) {
        int r = idx >> 7, j = idx & 127;
        float v = (float)X[r * 128 + j] + (float)Y[r * 128 + j];
        X[r * 128 + j] = (half_t)((v - stat[r]) * stat[16 + r] * g[j] + b[j]);
    }
    __syncthreads();
}

struct EncP {
    const float* tokens;
    const half_t* w_eda_proj; const float* b_eda_proj; const float* eda_cls;
    const half_t* w_imu_proj; const float* b_imu_proj; const float* imu_cls;
    const half_t* w_e2i_v[L_NUM]; const float* b_e2i_v[L_NUM];
    const half_t* w_e2i_o[L_NUM]; const float* b_e2i_o[L_NUM];
    const half_t* w_fe1[L_NUM];   const float* b_fe1[L_NUM];
    const half_t* w_fe2[L_NUM];   const float* b_fe2[L_NUM];
    const half_t* w_i2e_v[L_NUM]; const float* b_i2e_v[L_NUM];
    const half_t* w_i2e_o[L_NUM]; const float* b_i2e_o[L_NUM];
    const half_t* w_fi1[L_NUM];   const float* b_fi1[L_NUM];
    const half_t* w_fi2[L_NUM];   const float* b_fi2[L_NUM];
    const float* ln_e1g[L_NUM]; const float* ln_e1b[L_NUM];
    const float* ln_e2g[L_NUM]; const float* ln_e2b[L_NUM];
    const float* ln_i1g[L_NUM]; const float* ln_i1b[L_NUM];
    const float* ln_i2g[L_NUM]; const float* ln_i2b[L_NUM];
    const half_t* w_out; const float* b_out;
    const float* oln_g;  const float* oln_b;
};

// Weight pre-pass: fp32 (N,K) row-major  ->  f16 WMMA-B swizzled tiles in d_ws.
// dst[tile*512 + lane*16 + i] = W[n0 + lane&15][k0 + (lane>>4)*8 + (i&7) + (i>>3)*16],
// tiles k-major: tile = (n0/16)*(K/32) + (k0/32)
__global__ void conv_w_kernel(const float* __restrict__ W, half_t* __restrict__ dst,
                              int N, int K) {
    int idx = blockIdx.x * 256 + threadIdx.x;
    if (idx >= N * K) return;
    int tile = idx >> 9, within = idx & 511;
    int lane = within >> 4, i = within & 15;
    int tiles_k = K >> 5;
    int tn = tile / tiles_k, tk = tile - tn * tiles_k;
    int n = tn * 16 + (lane & 15);
    int k = tk * 32 + (lane >> 4) * 8 + (i & 7) + (i >> 3) * 16;
    dst[idx] = (half_t)W[(size_t)n * K + k];
}

// 8 waves/block, 16 samples/wave (128 samples/block): concurrent waves walk the
// same weight tiles together -> WGP$ de-duplicates the L2 weight traffic.
// Seq-len == 1 => softmax over one score == 1, so
// MHA(q,k) == Wo @ (Wv @ k + bv) + bo   (Q/K projections are dead code).
__global__ __launch_bounds__(32 * WAVES_PER_BLK)
void enc_fused_kernel(EncP P, float* __restrict__ out) {
    extern __shared__ char smem_raw[];
    const int wave = threadIdx.x >> 5;
    const int lane = threadIdx.x & 31;
    char* wb = smem_raw + wave * WAVE_BYTES;
    half_t* eda = (half_t*)(wb + OFF_EDA);
    half_t* imu = (half_t*)(wb + OFF_IMU);
    half_t* hh  = (half_t*)(wb + OFF_H);
    half_t* tt  = (half_t*)(wb + OFF_T);
    float*  st  = (float*)(wb + OFF_STAT);

    const size_t s0 = ((size_t)blockIdx.x * WAVES_PER_BLK + wave) * 16;

    // ---- input projections (+cls), tokens streamed once from HBM
    gemm_proj<D_IN>(P.tokens + s0 * 2 * D_IN,        2 * D_IN, P.w_eda_proj,
                    P.b_eda_proj, P.eda_cls, eda, lane);
    gemm_proj<D_IN>(P.tokens + s0 * 2 * D_IN + D_IN, 2 * D_IN, P.w_imu_proj,
                    P.b_imu_proj, P.imu_cls, imu, lane);
    __syncthreads();

#pragma unroll
    for (int l = 0; l < L_NUM; ++l) {
        // eda <- cross-attn(eda, imu)  (affine in imu)
        gemm_l<128, 128, false>(imu, 128, P.w_e2i_v[l], P.b_e2i_v[l], hh, 128, lane);
        __syncthreads();
        gemm_l<128, 128, false>(hh, 128, P.w_e2i_o[l], P.b_e2i_o[l], tt, 128, lane);
        __syncthreads();
        residual_ln16(eda, tt, P.ln_e1g[l], P.ln_e1b[l], st, lane);

        // eda FFN
        gemm_l<256, 128, true >(eda, 128, P.w_fe1[l], P.b_fe1[l], hh, 256, lane);
        __syncthreads();
        gemm_l<128, 256, false>(hh, 256, P.w_fe2[l], P.b_fe2[l], tt, 128, lane);
        __syncthreads();
        residual_ln16(eda, tt, P.ln_e2g[l], P.ln_e2b[l], st, lane);

        // imu <- cross-attn(imu, eda_updated)
        gemm_l<128, 128, false>(eda, 128, P.w_i2e_v[l], P.b_i2e_v[l], hh, 128, lane);
        __syncthreads();
        gemm_l<128, 128, false>(hh, 128, P.w_i2e_o[l], P.b_i2e_o[l], tt, 128, lane);
        __syncthreads();
        residual_ln16(imu, tt, P.ln_i1g[l], P.ln_i1b[l], st, lane);

        // imu FFN
        gemm_l<256, 128, true >(imu, 128, P.w_fi1[l], P.b_fi1[l], hh, 256, lane);
        __syncthreads();
        gemm_l<128, 256, false>(hh, 256, P.w_fi2[l], P.b_fi2[l], tt, 128, lane);
        __syncthreads();
        residual_ln16(imu, tt, P.ln_i2g[l], P.ln_i2b[l], st, lane);
    }

    // ---- head: out = relu(LN(concat(eda, imu) @ out_w^T + out_b))
    gemm_cat16(eda, imu, P.w_out, P.b_out, tt, lane);
    __syncthreads();

    if (lane < 16) {
        float s = 0.0f, s2 = 0.0f;
#pragma unroll 8
        for (int j = 0; j < 128; ++j) {
            float v = (float)tt[lane * 128 + j];
            s += v; s2 += v * v;
        }
        float m   = s * (1.0f / 128.0f);
        float var = s2 * (1.0f / 128.0f) - m * m;
        st[lane]      = m;
        st[16 + lane] = __frsqrt_rn(var + 1e-5f);
    }
    __syncthreads();
    for (int idx = lane; idx < 16 * 128; idx += 32) {
        int r = idx >> 7, j = idx & 127;
        float v = ((float)tt[r * 128 + j] - st[r]) * st[16 + r] * P.oln_g[j] + P.oln_b[j];
        out[(s0 + r) * 128 + j] = fmaxf(v, 0.0f);
    }
}

extern "C" void kernel_launch(void* const* d_in, const int* in_sizes, int n_in,
                              void* d_out, int out_size, void* d_ws, size_t ws_size,
                              hipStream_t stream) {
    (void)in_sizes; (void)n_in; (void)out_size; (void)ws_size;
    const float* f_tokens   = (const float*)d_in[0];
    const float* f_edapw    = (const float*)d_in[1];
    const float* f_edapb    = (const float*)d_in[2];
    const float* f_imupw    = (const float*)d_in[3];
    const float* f_imupb    = (const float*)d_in[4];
    const float* f_edacls   = (const float*)d_in[5];
    const float* f_imucls   = (const float*)d_in[6];
    const float* f_e2iwqkv  = (const float*)d_in[7];
    const float* f_e2ibqkv  = (const float*)d_in[8];
    const float* f_e2iwo    = (const float*)d_in[9];
    const float* f_e2ibo    = (const float*)d_in[10];
    const float* f_i2ewqkv  = (const float*)d_in[11];
    const float* f_i2ebqkv  = (const float*)d_in[12];
    const float* f_i2ewo    = (const float*)d_in[13];
    const float* f_i2ebo    = (const float*)d_in[14];
    const float* f_fe1      = (const float*)d_in[15];
    const float* f_fe1b     = (const float*)d_in[16];
    const float* f_fe2      = (const float*)d_in[17];
    const float* f_fe2b     = (const float*)d_in[18];
    const float* f_fi1      = (const float*)d_in[19];
    const float* f_fi1b     = (const float*)d_in[20];
    const float* f_fi2      = (const float*)d_in[21];
    const float* f_fi2b     = (const float*)d_in[22];

    half_t* ws = (half_t*)d_ws;
    size_t off = 0;
    auto conv = [&](const float* W, int N, int K) -> const half_t* {
        half_t* dst = ws + off;
        int total = N * K;
        conv_w_kernel<<<(total + 255) / 256, 256, 0, stream>>>(W, dst, N, K);
        off += (size_t)total;
        return dst;
    };

    EncP P;
    P.tokens     = f_tokens;
    P.w_eda_proj = conv(f_edapw, 128, 512);
    P.b_eda_proj = f_edapb;  P.eda_cls = f_edacls;
    P.w_imu_proj = conv(f_imupw, 128, 512);
    P.b_imu_proj = f_imupb;  P.imu_cls = f_imucls;

    for (int l = 0; l < L_NUM; ++l) {
        P.w_e2i_v[l] = conv(f_e2iwqkv + ((size_t)l * 384 + 256) * 128, 128, 128);
        P.b_e2i_v[l] = f_e2ibqkv + (size_t)l * 384 + 256;
        P.w_e2i_o[l] = conv(f_e2iwo + (size_t)l * 128 * 128, 128, 128);
        P.b_e2i_o[l] = f_e2ibo + (size_t)l * 128;
        P.w_fe1[l]   = conv(f_fe1 + (size_t)l * 256 * 128, 256, 128);
        P.b_fe1[l]   = f_fe1b + (size_t)l * 256;
        P.w_fe2[l]   = conv(f_fe2 + (size_t)l * 128 * 256, 128, 256);
        P.b_fe2[l]   = f_fe2b + (size_t)l * 128;
        P.w_i2e_v[l] = conv(f_i2ewqkv + ((size_t)l * 384 + 256) * 128, 128, 128);
        P.b_i2e_v[l] = f_i2ebqkv + (size_t)l * 384 + 256;
        P.w_i2e_o[l] = conv(f_i2ewo + (size_t)l * 128 * 128, 128, 128);
        P.b_i2e_o[l] = f_i2ebo + (size_t)l * 128;
        P.w_fi1[l]   = conv(f_fi1 + (size_t)l * 256 * 128, 256, 128);
        P.b_fi1[l]   = f_fi1b + (size_t)l * 256;
        P.w_fi2[l]   = conv(f_fi2 + (size_t)l * 128 * 256, 128, 256);
        P.b_fi2[l]   = f_fi2b + (size_t)l * 128;
        P.ln_e1g[l] = (const float*)d_in[23] + l * 128;
        P.ln_e1b[l] = (const float*)d_in[24] + l * 128;
        P.ln_e2g[l] = (const float*)d_in[25] + l * 128;
        P.ln_e2b[l] = (const float*)d_in[26] + l * 128;
        P.ln_i1g[l] = (const float*)d_in[27] + l * 128;
        P.ln_i1b[l] = (const float*)d_in[28] + l * 128;
        P.ln_i2g[l] = (const float*)d_in[29] + l * 128;
        P.ln_i2b[l] = (const float*)d_in[30] + l * 128;
    }
    P.w_out = conv((const float*)d_in[31], 128, 256);
    P.b_out = (const float*)d_in[32];
    P.oln_g = (const float*)d_in[33];
    P.oln_b = (const float*)d_in[34];

    static bool attr_set = false;  // idempotent attribute set (host-side only)
    (void)attr_set;
    hipFuncSetAttribute((const void*)enc_fused_kernel,
                        hipFuncAttributeMaxDynamicSharedMemorySize, SMEM_BYTES);

    dim3 grid(B_TOT / (16 * WAVES_PER_BLK));
    dim3 block(32 * WAVES_PER_BLK);
    enc_fused_kernel<<<grid, block, SMEM_BYTES, stream>>>(P, (float*)d_out);
}
